// DeepseekV2AttentionMLA_8048768713518
// MI455X (gfx1250) — matrix operand order repaired and checked by reading, not compiled, not run on hardware
//
#include <hip/hip_runtime.h>
#include <hip/hip_bf16.h>

// ---------------- problem constants ----------------
#define T_SEQ 2048
#define HID   5120
#define QLR   1536
#define KVLR  512
#define NOPE  128
#define ROPE_D 64
#define VDIM  128
#define NH    32
#define QKD   192              // NOPE + ROPE
#define DQK   576              // KVLR + ROPE
#define SM_SCALE 0.07216878364870323f   // 192^-0.5
#define RMS_EPS 1e-6f

// ---------------- WMMA fragment types (CDNA5 wave32) ----------------
typedef __attribute__((ext_vector_type(16))) __bf16   v16bf;
typedef __attribute__((ext_vector_type(8)))  float    v8f;
typedef __attribute__((ext_vector_type(8)))  unsigned v8u;

union ABfrag { v8u u; v16bf bf; };

__device__ __forceinline__ unsigned short f2bf(float f) {
  unsigned u = __float_as_uint(f);
  unsigned r = u + 0x7FFFu + ((u >> 16) & 1u);   // round-to-nearest-even
  return (unsigned short)(r >> 16);
}

// ---- CDNA5 async global->LDS copy (ASYNCcnt-tracked), 16B per lane ----
// CDNA5 ISA: GLOBAL_LOAD_ASYNC_TO_LDS_B128 (GV mode).
// lds_dst low 32 bits = segment-relative LDS byte offset (HW adds LDS_BASE).
__device__ __forceinline__ void async_b128(void* lds_dst, const void* gsrc) {
  unsigned loff = (unsigned)(unsigned long long)lds_dst;
  unsigned long long ga = (unsigned long long)gsrc;
  asm volatile("global_load_async_to_lds_b128 %0, %1, off"
               :: "v"(loff), "v"(ga) : "memory");
}
__device__ __forceinline__ void wait_async0() {
  asm volatile("s_wait_asynccnt 0x0" ::: "memory");
}

// ---------------- fp32 -> bf16 convert (4 elems / thread) ----------------
__global__ void cvt_f32_bf16(const float* __restrict__ in,
                             unsigned short* __restrict__ out, long long n4) {
  long long i = blockIdx.x * (long long)blockDim.x + threadIdx.x;
  if (i >= n4) return;
  float4 f = ((const float4*)in)[i];
  unsigned lo = (unsigned)f2bf(f.x) | ((unsigned)f2bf(f.y) << 16);
  unsigned hi = (unsigned)f2bf(f.z) | ((unsigned)f2bf(f.w) << 16);
  ((uint2*)out)[i] = make_uint2(lo, hi);
}

// ---------------- RoPE cos/sin tables ----------------
__global__ void rope_tables(const long long* __restrict__ pos,
                            float* __restrict__ cosT, float* __restrict__ sinT) {
  int t = blockIdx.x;                 // T blocks
  int i = threadIdx.x;                // 32 pairs
  float inv = powf(10000.0f, -(float)(2 * i) / 64.0f);
  float f = (float)pos[t] * inv;
  cosT[t * 32 + i] = cosf(f);
  sinT[t * 32 + i] = sinf(f);
}

// ---------------- rmsnorm: fp32 row -> bf16 row ----------------
__global__ __launch_bounds__(256)
void rmsnorm_bf16(const float* __restrict__ in, const float* __restrict__ wt,
                  unsigned short* __restrict__ out, int cols) {
  __shared__ float red[8];
  __shared__ float srs;
  int row = blockIdx.x, tid = threadIdx.x;
  const float* x = in + (long long)row * cols;
  float s = 0.f;
  for (int c = tid; c < cols; c += 256) { float v = x[c]; s += v * v; }
#pragma unroll
  for (int o = 16; o; o >>= 1) s += __shfl_xor(s, o, 32);
  if ((tid & 31) == 0) red[tid >> 5] = s;
  __syncthreads();
  if (tid == 0) {
    float tot = 0.f;
    for (int i = 0; i < 8; ++i) tot += red[i];
    srs = rsqrtf(tot / (float)cols + RMS_EPS);
  }
  __syncthreads();
  float rs = srs;
  for (int c = tid; c < cols; c += 256)
    out[(long long)row * cols + c] = f2bf(x[c] * rs * wt[c]);
}

// ---------------- kv latent -> v_input (bf16) + k_input (bf16, rmsnorm||rope) ----------------
__global__ __launch_bounds__(128)
void kv_prep(const float* __restrict__ lat, const float* __restrict__ lnw,
             const float* __restrict__ cosT, const float* __restrict__ sinT,
             unsigned short* __restrict__ v_bf, unsigned short* __restrict__ k_bf) {
  __shared__ float red[4];
  __shared__ float srs;
  int t = blockIdx.x, tid = threadIdx.x;
  const float* x = lat + (long long)t * DQK;
  float s = 0.f;
  for (int c = tid; c < KVLR; c += 128) { float v = x[c]; s += v * v; }
#pragma unroll
  for (int o = 16; o; o >>= 1) s += __shfl_xor(s, o, 32);
  if ((tid & 31) == 0) red[tid >> 5] = s;
  __syncthreads();
  if (tid == 0) {
    float tot = red[0] + red[1] + red[2] + red[3];
    srs = rsqrtf(tot / (float)KVLR + RMS_EPS);
  }
  __syncthreads();
  float rs = srs;
  for (int c = tid; c < KVLR; c += 128) {
    unsigned short b = f2bf(x[c] * rs * lnw[c]);
    v_bf[(long long)t * KVLR + c] = b;
    k_bf[(long long)t * DQK + c]  = b;
  }
  if (tid < 32) {
    int i = tid;
    float x1 = x[KVLR + 2 * i], x2 = x[KVLR + 2 * i + 1];
    float c = cosT[t * 32 + i], sn = sinT[t * 32 + i];
    k_bf[(long long)t * DQK + KVLR + 2 * i]     = f2bf(x1 * c - x2 * sn);
    k_bf[(long long)t * DQK + KVLR + 2 * i + 1] = f2bf(x2 * c + x1 * sn);
  }
}

// ---------------- rope on q_pe -> q_input[..., 512:576] ----------------
__global__ void q_rope(const float* __restrict__ q, const float* __restrict__ cosT,
                       const float* __restrict__ sinT, unsigned short* __restrict__ qin) {
  int t = blockIdx.x, h = blockIdx.y, i = threadIdx.x;   // 32 pairs
  const float* src = q + (long long)t * (NH * QKD) + h * QKD + NOPE;
  float x1 = src[2 * i], x2 = src[2 * i + 1];
  float c = cosT[t * 32 + i], s = sinT[t * 32 + i];
  unsigned short* dst = qin + ((long long)t * NH + h) * DQK + KVLR;
  dst[2 * i]     = f2bf(x1 * c - x2 * s);
  dst[2 * i + 1] = f2bf(x2 * c + x1 * s);
}

// ---------------- generic batched bf16 WMMA GEMM: C(64x64 tile) = A(MxK) @ B(KxN) ----------------
// A row-major bf16 (async-staged to LDS), B row-major bf16 (staged transposed in LDS),
// C fp32 or bf16. M,N multiples of 64; K multiple of 32 (guaranteed by caller).
#define LA_STRIDE 40   // u16; 80B row pitch -> 16B-aligned async B128 destinations

template <bool OUT_BF16>
__global__ __launch_bounds__(256)
void gemm_bf16(const unsigned short* __restrict__ A, int lda, long long sA,
               const unsigned short* __restrict__ B, int ldb, long long sB,
               void* __restrict__ Cv, int ldc, long long sC, int K) {
  __shared__ unsigned short ldsA[64][LA_STRIDE];   // [m][k]
  __shared__ unsigned short ldsB[64][34];          // transposed: [n][k]
  A += (long long)blockIdx.z * sA;
  B += (long long)blockIdx.z * sB;
  int row0 = blockIdx.y * 64, col0 = blockIdx.x * 64;
  int tid = threadIdx.x;
  int lane = tid & 31, w = tid >> 5;
  int mi = w & 3, nh = w >> 2;              // 4 m-frags x 2 n-halves
  int g = lane >> 4, l16 = lane & 15;

  v8f acc0 = {0.f, 0.f, 0.f, 0.f, 0.f, 0.f, 0.f, 0.f};
  v8f acc1 = acc0;

  for (int k0 = 0; k0 < K; k0 += 32) {
    __syncthreads();
    {   // A tile 64x32: async global->LDS, 16B (8 bf16) per thread
      int r = tid >> 2, c = (tid & 3) * 8;
      async_b128(&ldsA[r][c], A + (long long)(row0 + r) * lda + k0 + c);
    }
    {   // B tile 32x64 -> transposed LDS (manual: transpose in regs)
      int kk = tid >> 3, n0 = (tid & 7) * 8;
      uint4 d = *(const uint4*)(B + (long long)(k0 + kk) * ldb + col0 + n0);
      unsigned short tmp[8];
      *(uint4*)tmp = d;
#pragma unroll
      for (int j = 0; j < 8; ++j) ldsB[n0 + j][kk] = tmp[j];
    }
    wait_async0();
    __syncthreads();

    ABfrag a, b0, b1;
#pragma unroll
    for (int v = 0; v < 8; ++v) {
      int ka = 2 * v + 16 * g;                          // contiguous k = 16g + s for A
      a.u[v]  = *(const unsigned*)&ldsA[mi * 16 + l16][ka];
      int kb = 2 * v + 16 * g;                          // contiguous k = 16g + s for B, same as A
      b0.u[v] = *(const unsigned*)&ldsB[nh * 32 + l16][kb];
      b1.u[v] = *(const unsigned*)&ldsB[nh * 32 + 16 + l16][kb];
    }
    acc0 = __builtin_amdgcn_wmma_f32_16x16x32_bf16(false, a.bf, false, b0.bf,
                                                   (short)0, acc0, false, false);
    acc1 = __builtin_amdgcn_wmma_f32_16x16x32_bf16(false, a.bf, false, b1.bf,
                                                   (short)0, acc1, false, false);
  }

  int r_base = row0 + mi * 16 + 8 * g;
  int c_base = col0 + nh * 32 + l16;
  if (OUT_BF16) {
    unsigned short* C = (unsigned short*)Cv + (long long)blockIdx.z * sC;
#pragma unroll
    for (int v = 0; v < 8; ++v) {
      C[(long long)(r_base + v) * ldc + c_base]      = f2bf(acc0[v]);
      C[(long long)(r_base + v) * ldc + c_base + 16] = f2bf(acc1[v]);
    }
  } else {
    float* C = (float*)Cv + (long long)blockIdx.z * sC;
#pragma unroll
    for (int v = 0; v < 8; ++v) {
      C[(long long)(r_base + v) * ldc + c_base]      = acc0[v];
      C[(long long)(r_base + v) * ldc + c_base + 16] = acc1[v];
    }
  }
}

// ---------------- flash attention (per head, 16 query rows / block, 32 keys / iter) ----------------
// qin: (T, H, 576) bf16 ; kin: (T, 576) bf16 ; vin: (T, 512) bf16 ; ctx: (T, H, 512) bf16
#define SQ_STRIDE 584     // 576 + 8 pad (u16); 1168B row pitch -> 16B-aligned async dests
#define SV_STRIDE 34      // 32 keys + 2 pad (u16)
#define SP_STRIDE 34

__global__ __launch_bounds__(128)
void flash_attn(const unsigned short* __restrict__ qin,
                const unsigned short* __restrict__ kin,
                const unsigned short* __restrict__ vin,
                unsigned short* __restrict__ ctx) {
  extern __shared__ unsigned short smem[];
  unsigned short* sQ  = smem;                         // [16][584]
  unsigned short* sK  = sQ + 16 * SQ_STRIDE;          // [32][584]
  unsigned short* sVt = sK + 32 * SQ_STRIDE;          // [512][34]  (transposed V)
  unsigned short* sP  = sVt + 512 * SV_STRIDE;        // [16][34]
  float* sAlpha = (float*)(sP + 16 * SP_STRIDE);      // [16]
  float* sL     = sAlpha + 16;                        // [16]

  int m0 = blockIdx.x * 16;
  int h  = blockIdx.y;
  int tid = threadIdx.x;
  int lane = tid & 31, w = tid >> 5;                  // 4 waves
  int g = lane >> 4, l16 = lane & 15;

  // stage Q tile (16 x 576) asynchronously
  for (int i = tid; i < 16 * 72; i += 128) {
    int r = i / 72, c = (i % 72) * 8;
    async_b128(&sQ[r * SQ_STRIDE + c],
               qin + ((long long)(m0 + r) * NH + h) * DQK + c);
  }
  wait_async0();

  v8f zf = {0.f, 0.f, 0.f, 0.f, 0.f, 0.f, 0.f, 0.f};
  v8f acc[8];
#pragma unroll
  for (int f = 0; f < 8; ++f) acc[f] = zf;
  float m_run[8], l_run[8];
#pragma unroll
  for (int v = 0; v < 8; ++v) { m_run[v] = -1e30f; l_run[v] = 0.f; }

  int ktmax = (m0 + 15) >> 5;
  for (int kt = 0; kt <= ktmax; ++kt) {
    int s0 = kt * 32;
    __syncthreads();
    // stage K tile (32 x 576) asynchronously
    for (int i = tid; i < 32 * 72; i += 128) {
      int r = i / 72, c = (i % 72) * 8;
      async_b128(&sK[r * SQ_STRIDE + c],
                 kin + (long long)(s0 + r) * DQK + c);
    }
    // stage V tile transposed (512 vdims x 32 keys) — manual transpose
    for (int i = tid; i < 32 * 64; i += 128) {
      int key = i >> 6, v0 = (i & 63) * 8;
      uint4 d = *(const uint4*)(vin + (long long)(s0 + key) * KVLR + v0);
      unsigned short tmp[8];
      *(uint4*)tmp = d;
#pragma unroll
      for (int j = 0; j < 8; ++j) sVt[(v0 + j) * SV_STRIDE + key] = tmp[j];
    }
    wait_async0();
    __syncthreads();

    if (w == 0) {   // wave 0: scores + online softmax
      v8f sc0 = zf, sc1 = zf;
#pragma unroll 2
      for (int ks = 0; ks < 18; ++ks) {
        ABfrag a, b0, b1;
#pragma unroll
        for (int v = 0; v < 8; ++v) {
          int ka = ks * 32 + 2 * v + 16 * g;
          a.u[v]  = *(const unsigned*)&sQ[l16 * SQ_STRIDE + ka];
          int kb = ks * 32 + 2 * v + 16 * g;
          b0.u[v] = *(const unsigned*)&sK[l16 * SQ_STRIDE + kb];
          b1.u[v] = *(const unsigned*)&sK[(16 + l16) * SQ_STRIDE + kb];
        }
        sc0 = __builtin_amdgcn_wmma_f32_16x16x32_bf16(false, a.bf, false, b0.bf,
                                                      (short)0, sc0, false, false);
        sc1 = __builtin_amdgcn_wmma_f32_16x16x32_bf16(false, a.bf, false, b1.bf,
                                                      (short)0, sc1, false, false);
      }
#pragma unroll
      for (int v = 0; v < 8; ++v) {
        int r = v + 8 * g;
        int t = m0 + r;
        float e0 = sc0[v] * SM_SCALE;
        float e1 = sc1[v] * SM_SCALE;
        if (s0 + l16 > t)      e0 = -1e30f;
        if (s0 + 16 + l16 > t) e1 = -1e30f;
        float mx = fmaxf(e0, e1);
#pragma unroll
        for (int o = 1; o < 16; o <<= 1) mx = fmaxf(mx, __shfl_xor(mx, o, 16));
        float mnew = fmaxf(m_run[v], mx);
        float al = expf(m_run[v] - mnew);
        float p0 = expf(e0 - mnew), p1 = expf(e1 - mnew);
        float rs = p0 + p1;
#pragma unroll
        for (int o = 1; o < 16; o <<= 1) rs += __shfl_xor(rs, o, 16);
        l_run[v] = l_run[v] * al + rs;
        m_run[v] = mnew;
        if (l16 == 0) sAlpha[r] = al;
        sP[r * SP_STRIDE + l16]      = f2bf(p0);
        sP[r * SP_STRIDE + 16 + l16] = f2bf(p1);
      }
    }
    __syncthreads();

    // all waves: rescale accumulators and P(16x32) @ V(32x128-slice)
    float al[8];
#pragma unroll
    for (int v = 0; v < 8; ++v) al[v] = sAlpha[v + 8 * g];
    ABfrag pf;
#pragma unroll
    for (int v = 0; v < 8; ++v) {
      int ka = 2 * v + 16 * g;
      pf.u[v] = *(const unsigned*)&sP[l16 * SP_STRIDE + ka];
    }
#pragma unroll
    for (int f = 0; f < 8; ++f) {
      ABfrag bv;
#pragma unroll
      for (int v = 0; v < 8; ++v) {
        int kb = 2 * v + 16 * g;
        bv.u[v] = *(const unsigned*)&sVt[(w * 128 + f * 16 + l16) * SV_STRIDE + kb];
      }
      v8f t = acc[f];
#pragma unroll
      for (int v = 0; v < 8; ++v) t[v] *= al[v];
      acc[f] = __builtin_amdgcn_wmma_f32_16x16x32_bf16(false, pf.bf, false, bv.bf,
                                                       (short)0, t, false, false);
    }
  }

  if (w == 0 && l16 == 0) {
#pragma unroll
    for (int v = 0; v < 8; ++v) sL[v + 8 * g] = l_run[v];
  }
  __syncthreads();
  float rl[8];
#pragma unroll
  for (int v = 0; v < 8; ++v) rl[v] = 1.0f / sL[v + 8 * g];
#pragma unroll
  for (int f = 0; f < 8; ++f) {
#pragma unroll
    for (int v = 0; v < 8; ++v) {
      int r = m0 + v + 8 * g;
      int c = w * 128 + f * 16 + l16;
      ctx[((long long)r * NH + h) * KVLR + c] = f2bf(acc[f][v] * rl[v]);
    }
  }
}

// ---------------- host-side orchestration ----------------
extern "C" void kernel_launch(void* const* d_in, const int* in_sizes, int n_in,
                              void* d_out, int out_size, void* d_ws, size_t ws_size,
                              hipStream_t stream) {
  (void)in_sizes; (void)n_in; (void)out_size; (void)ws_size;

  const float*      hidden    = (const float*)d_in[0];
  const long long*  positions = (const long long*)d_in[1];
  const float*      w_q_a     = (const float*)d_in[2];
  const float*      q_a_ln_w  = (const float*)d_in[3];
  const float*      w_q_b     = (const float*)d_in[4];
  const float*      w_kv_a    = (const float*)d_in[5];
  const float*      kv_a_ln_w = (const float*)d_in[6];
  const float*      w_kc      = (const float*)d_in[7];
  const float*      w_vc      = (const float*)d_in[8];
  const float*      w_o       = (const float*)d_in[9];
  float*            out       = (float*)d_out;

  // workspace carve (256B aligned)
  size_t off = 0;
  auto carve = [&](size_t bytes) -> void* {
    off = (off + 255) & ~(size_t)255;
    void* p = (char*)d_ws + off;
    off += bytes;
    return p;
  };
  unsigned short* hid_bf  = (unsigned short*)carve((size_t)T_SEQ * HID * 2);
  unsigned short* wqa_bf  = (unsigned short*)carve((size_t)HID * QLR * 2);
  unsigned short* wqb_bf  = (unsigned short*)carve((size_t)QLR * NH * QKD * 2);
  unsigned short* wkva_bf = (unsigned short*)carve((size_t)HID * DQK * 2);
  unsigned short* wkc_bf  = (unsigned short*)carve((size_t)NH * NOPE * KVLR * 2);
  unsigned short* wvc_bf  = (unsigned short*)carve((size_t)NH * KVLR * VDIM * 2);
  unsigned short* wo_bf   = (unsigned short*)carve((size_t)NH * VDIM * HID * 2);
  float*          qa_f    = (float*)carve((size_t)T_SEQ * QLR * 4);
  unsigned short* qan_bf  = (unsigned short*)carve((size_t)T_SEQ * QLR * 2);
  float*          q_f     = (float*)carve((size_t)T_SEQ * NH * QKD * 4);
  unsigned short* q_bf    = (unsigned short*)carve((size_t)T_SEQ * NH * QKD * 2);
  float*          lat_f   = (float*)carve((size_t)T_SEQ * DQK * 4);
  unsigned short* v_bf    = (unsigned short*)carve((size_t)T_SEQ * KVLR * 2);
  unsigned short* k_bf    = (unsigned short*)carve((size_t)T_SEQ * DQK * 2);
  unsigned short* qin_bf  = (unsigned short*)carve((size_t)T_SEQ * NH * DQK * 2);
  float*          cosT    = (float*)carve((size_t)T_SEQ * 32 * 4);
  float*          sinT    = (float*)carve((size_t)T_SEQ * 32 * 4);
  unsigned short* ctx_bf  = (unsigned short*)carve((size_t)T_SEQ * NH * KVLR * 2);
  float*          attn_f  = (float*)carve((size_t)T_SEQ * NH * VDIM * 4);
  unsigned short* attn_bf = (unsigned short*)carve((size_t)T_SEQ * NH * VDIM * 2);

  auto cvt = [&](const float* src, unsigned short* dst, long long n) {
    long long n4 = n / 4;
    int blk = 256;
    long long grid = (n4 + blk - 1) / blk;
    cvt_f32_bf16<<<(unsigned)grid, blk, 0, stream>>>(src, dst, n4);
  };

  // weight / activation conversions to bf16
  cvt(hidden, hid_bf, (long long)T_SEQ * HID);
  cvt(w_q_a,  wqa_bf, (long long)HID * QLR);
  cvt(w_q_b,  wqb_bf, (long long)QLR * NH * QKD);
  cvt(w_kv_a, wkva_bf,(long long)HID * DQK);
  cvt(w_kc,   wkc_bf, (long long)NH * NOPE * KVLR);
  cvt(w_vc,   wvc_bf, (long long)NH * KVLR * VDIM);
  cvt(w_o,    wo_bf,  (long long)NH * VDIM * HID);

  rope_tables<<<T_SEQ, 32, 0, stream>>>(positions, cosT, sinT);

  // q_a = hidden @ w_q_a   (2048x1536, K=5120)
  gemm_bf16<false><<<dim3(QLR / 64, T_SEQ / 64, 1), 256, 0, stream>>>(
      hid_bf, HID, 0, wqa_bf, QLR, 0, qa_f, QLR, 0, HID);
  // rmsnorm(q_a)
  rmsnorm_bf16<<<T_SEQ, 256, 0, stream>>>(qa_f, q_a_ln_w, qan_bf, QLR);
  // q = qan @ w_q_b   (2048x6144, K=1536)
  gemm_bf16<false><<<dim3(NH * QKD / 64, T_SEQ / 64, 1), 256, 0, stream>>>(
      qan_bf, QLR, 0, wqb_bf, NH * QKD, 0, q_f, NH * QKD, 0, QLR);
  cvt(q_f, q_bf, (long long)T_SEQ * NH * QKD);

  // latent = hidden @ w_kv_a   (2048x576, K=5120)
  gemm_bf16<false><<<dim3(DQK / 64, T_SEQ / 64, 1), 256, 0, stream>>>(
      hid_bf, HID, 0, wkva_bf, DQK, 0, lat_f, DQK, 0, HID);
  kv_prep<<<T_SEQ, 128, 0, stream>>>(lat_f, kv_a_ln_w, cosT, sinT, v_bf, k_bf);

  // q_nope_out[h] = q_nope[h] @ w_kc[h]  ->  q_input[:, h, 0:512]  (bf16 out, batched over H)
  gemm_bf16<true><<<dim3(KVLR / 64, T_SEQ / 64, NH), 256, 0, stream>>>(
      q_bf, NH * QKD, (long long)QKD,
      wkc_bf, KVLR, (long long)NOPE * KVLR,
      qin_bf, NH * DQK, (long long)DQK, NOPE);
  // q_pe rope -> q_input[:, h, 512:576]
  q_rope<<<dim3(T_SEQ, NH), 32, 0, stream>>>(q_f, cosT, sinT, qin_bf);

  // flash attention -> ctx (T, H, 512) bf16
  size_t flash_lds =
      (size_t)(16 * SQ_STRIDE + 32 * SQ_STRIDE + 512 * SV_STRIDE + 16 * SP_STRIDE) * 2 +
      32 * 4;
  (void)hipFuncSetAttribute((const void*)flash_attn,
                            hipFuncAttributeMaxDynamicSharedMemorySize,
                            (int)flash_lds);
  flash_attn<<<dim3(T_SEQ / 16, NH), 128, flash_lds, stream>>>(qin_bf, k_bf, v_bf, ctx_bf);

  // attn[h] = ctx[h] @ w_vc[h]   (2048x128, K=512, batched over H)
  gemm_bf16<false><<<dim3(VDIM / 64, T_SEQ / 64, NH), 256, 0, stream>>>(
      ctx_bf, NH * KVLR, (long long)KVLR,
      wvc_bf, VDIM, (long long)KVLR * VDIM,
      attn_f, NH * VDIM, (long long)VDIM, KVLR);
  cvt(attn_f, attn_bf, (long long)T_SEQ * NH * VDIM);

  // out = attn @ w_o   (2048x5120, K=4096)
  gemm_bf16<false><<<dim3(HID / 64, T_SEQ / 64, 1), 256, 0, stream>>>(
      attn_bf, NH * VDIM, 0, wo_bf, HID, 0, out, HID, 0, NH * VDIM);
}
